// HeteroClassifier_87857851007505
// MI455X (gfx1250) — matrix operand order repaired and checked by reading, not compiled, not run on hardware
//
#include <hip/hip_runtime.h>
#include <cstdint>

#define N_SENT   100000
#define N_CTXN   50000
#define DH       128
#define N_GRAPHS 32

typedef __attribute__((ext_vector_type(16))) __bf16          v16bf;
typedef __attribute__((ext_vector_type(16))) unsigned short  v16u;
typedef __attribute__((ext_vector_type(8)))  unsigned short  v8u;
typedef __attribute__((ext_vector_type(8)))  float           v8f;

// ---------- small utility kernels ----------

__global__ void fill_f32(float* __restrict__ p, float v, int n) {
  int i = blockIdx.x * blockDim.x + threadIdx.x;
  if (i < n) p[i] = v;
}

__global__ void zero_i32(int* __restrict__ p, int n) {
  int i = blockIdx.x * blockDim.x + threadIdx.x;
  if (i < n) p[i] = 0;
}

__global__ void count_deg(const int* __restrict__ src, const int* __restrict__ dst,
                          int* __restrict__ dsrc, int* __restrict__ ddst, int ne) {
  int e = blockIdx.x * blockDim.x + threadIdx.x;
  if (e < ne) {
    atomicAdd(&dsrc[src[e]], 1);
    atomicAdd(&ddst[dst[e]], 1);
  }
}

__global__ void deg_to_rsqrt(const int* __restrict__ deg, float* __restrict__ rs, int n) {
  int i = blockIdx.x * blockDim.x + threadIdx.x;
  if (i < n) {
    int d = deg[i];
    if (d < 1) d = 1;
    rs[i] = rsqrtf((float)d);
  }
}

// round-to-nearest-even float -> bf16 bits
__device__ __forceinline__ unsigned short f2bf(float f) {
  unsigned u = __float_as_uint(f);
  u += 0x7fffu + ((u >> 16) & 1u);
  return (unsigned short)(u >> 16);
}

// ---------- SpMM scatter: agg[dst] += x[src] * rs_out[src] ----------
// 32 lanes per edge, float4 per lane (128 cols).
__global__ __launch_bounds__(256)
void spmm_scatter(const float* __restrict__ x, const int* __restrict__ src,
                  const int* __restrict__ dst, const float* __restrict__ rs_out,
                  float* __restrict__ agg, int ne) {
  int t = blockIdx.x * blockDim.x + threadIdx.x;
  int e = t >> 5, j = t & 31;
  if (e >= ne) return;
  int s = src[e], d = dst[e];
  float sc = rs_out[s];
  float4 v = ((const float4*)(x + (size_t)s * DH))[j];
  float* o = agg + (size_t)d * DH + j * 4;
  atomicAdd(o + 0, v.x * sc);
  atomicAdd(o + 1, v.y * sc);
  atomicAdd(o + 2, v.z * sc);
  atomicAdd(o + 3, v.w * sc);
}

// ---------- pack W (128x128 f32, row-major) into wave32 B-fragment bf16 layout ----------
// B (KxN tile): lane = half*16 + (n%16); elements j=0..15 <-> K = 16*half + j (per kb block)
// Wp[((t*4+kb)*32 + lane)*16 + j] = bf16( W[(kb*32 + 16*half + j)*128 + (t*16 + lane%16)] )
__global__ void pack_w_bf16(const float* __restrict__ W, unsigned short* __restrict__ Wp) {
  int idx = blockIdx.x * blockDim.x + threadIdx.x;  // 16384 total
  if (idx >= 8 * 4 * 32 * 16) return;
  int j  = idx & 15;
  int L  = (idx >> 4) & 31;
  int kb = (idx >> 9) & 3;
  int t  = idx >> 11;
  int half = L >> 4;
  int n    = t * 16 + (L & 15);
  int k    = kb * 32 + 16 * half + j;
  Wp[idx] = f2bf(W[k * DH + n]);
}

// ---------- GEMM: out = RELU?( [ACC?out] + (agg*rs_in_row) @ Wbf16 + bias ) ----------
// Block: 256 threads = 8 waves; block owns a 16-row strip, wave w owns N-tile w.
// A strip is staged once through LDS (scaled + converted to bf16), then each wave
// reads its WMMA A fragments with ds_load_b128 pairs. ACC/RELU are compile-time.
#define A_STRIDE 136   // 128 + 8 halves pad -> bank-spread fragment reads

template <int ACC, int RELU>
__global__ __launch_bounds__(256)
void gemm_wmma(const float* __restrict__ agg, const float* __restrict__ rs_in,
               const unsigned short* __restrict__ Wp, const float* __restrict__ bias,
               float* __restrict__ out) {
  __shared__ unsigned short Abuf[16 * A_STRIDE];

  int tid = threadIdx.x;
  int m0  = blockIdx.x * 16;

  // ---- cooperative stage: 16 rows x 128 cols, scale by rs_in[row], cvt bf16 ----
  {
    int r  = tid >> 4;        // 0..15
    int cc = tid & 15;        // 8-col chunk
    int c  = cc * 8;
    int row = m0 + r;
    float sc = rs_in[row];
    const float* ap = agg + (size_t)row * DH + c;
    float4 f0 = *(const float4*)(ap);
    float4 f1 = *(const float4*)(ap + 4);
    v8u pk;
    pk[0] = f2bf(f0.x * sc); pk[1] = f2bf(f0.y * sc);
    pk[2] = f2bf(f0.z * sc); pk[3] = f2bf(f0.w * sc);
    pk[4] = f2bf(f1.x * sc); pk[5] = f2bf(f1.y * sc);
    pk[6] = f2bf(f1.z * sc); pk[7] = f2bf(f1.w * sc);
    *(v8u*)(Abuf + r * A_STRIDE + c) = pk;
  }
  __syncthreads();

  int lane = tid & 31;
  int t    = tid >> 5;       // N tile 0..7
  int half = lane >> 4;
  int lm   = lane & 15;

  v8f acc = {};
#pragma unroll
  for (int kb = 0; kb < 4; ++kb) {
    // A fragment (contiguous k order, matching Wp): elements 0..15 = K(kb*32 + 16*half + 0..15)
    const unsigned short* abase = Abuf + lm * A_STRIDE + kb * 32 + 16 * half;
    union { struct { v8u lo, hi; } p; v16bf b; } A;
    A.p.lo = *(const v8u*)(abase);
    A.p.hi = *(const v8u*)(abase + 8);

    union { v16u u; v16bf b; } B;
    B.u = *(const v16u*)(Wp + (size_t)(((t * 4 + kb) * 32 + lane) * 16));

    acc = __builtin_amdgcn_wmma_f32_16x16x32_bf16(false, A.b, false, B.b,
                                                  (short)0, acc, false, false);
  }

  int col = t * 16 + lm;
  float bv = bias[col];
  float* obase = out + (size_t)(m0 + 8 * half) * DH + col;

  float old[8];
  if (ACC) {
#pragma unroll
    for (int r = 0; r < 8; ++r) old[r] = obase[(size_t)r * DH];
  }
#pragma unroll
  for (int r = 0; r < 8; ++r) {
    float v = acc[r] + bv;
    if (ACC)  v += old[r];
    if (RELU) v = fmaxf(v, 0.0f);
    obase[(size_t)r * DH] = v;
  }
}

// ---------- segment max ----------

__device__ __forceinline__ void atomicMaxF(float* addr, float v) {
  if (v >= 0.0f) atomicMax((int*)addr, __float_as_int(v));
  else           atomicMin((unsigned int*)addr, __float_as_uint(v));
}

__global__ __launch_bounds__(256)
void segmax_kernel(const float* __restrict__ h, const int* __restrict__ gid,
                   float* __restrict__ out, int nrows) {
  int idx = blockIdx.x * blockDim.x + threadIdx.x;
  if (idx >= nrows * DH) return;
  int i = idx >> 7, c = idx & 127;
  atomicMaxF(&out[gid[i] * DH + c], h[idx]);
}

// ---------- host launch ----------

extern "C" void kernel_launch(void* const* d_in, const int* in_sizes, int n_in,
                              void* d_out, int out_size, void* d_ws, size_t ws_size,
                              hipStream_t stream) {
  const float* feat_s = (const float*)d_in[0];
  const float* feat_c = (const float*)d_in[1];

  const float *W1_s2s, *b1_s2s, *W2_s2s, *b2_s2s;
  const float *W1_s2c, *b1_s2c, *W2_s2c, *b2_s2c;
  const float *W1_c2s, *b1_c2s, *W2_c2s, *b2_c2s;
  if (in_sizes[3] == DH) {  // insertion order: (W1,b1,W2,b2) per rel s2s,s2c,c2s
    W1_s2s = (const float*)d_in[2];  b1_s2s = (const float*)d_in[3];
    W2_s2s = (const float*)d_in[4];  b2_s2s = (const float*)d_in[5];
    W1_s2c = (const float*)d_in[6];  b1_s2c = (const float*)d_in[7];
    W2_s2c = (const float*)d_in[8];  b2_s2c = (const float*)d_in[9];
    W1_c2s = (const float*)d_in[10]; b1_c2s = (const float*)d_in[11];
    W2_c2s = (const float*)d_in[12]; b2_c2s = (const float*)d_in[13];
  } else {                  // jax pytree sorted-key order
    W1_c2s = (const float*)d_in[2];  W1_s2c = (const float*)d_in[3];  W1_s2s = (const float*)d_in[4];
    W2_c2s = (const float*)d_in[5];  W2_s2c = (const float*)d_in[6];  W2_s2s = (const float*)d_in[7];
    b1_c2s = (const float*)d_in[8];  b1_s2c = (const float*)d_in[9];  b1_s2s = (const float*)d_in[10];
    b2_c2s = (const float*)d_in[11]; b2_s2c = (const float*)d_in[12]; b2_s2s = (const float*)d_in[13];
  }

  const int* e_s2s_src = (const int*)d_in[14];
  const int* e_s2s_dst = (const int*)d_in[15];
  const int* e_s2c_src = (const int*)d_in[16];
  const int* e_s2c_dst = (const int*)d_in[17];
  const int* e_c2s_src = (const int*)d_in[18];
  const int* e_c2s_dst = (const int*)d_in[19];
  const int* gid       = (const int*)d_in[20];
  int ne_s2s = in_sizes[14], ne_s2c = in_sizes[16], ne_c2s = in_sizes[18];

  // ---- workspace carve ----
  char* p = (char*)d_ws;
  int* deg = (int*)p;
  const int DEG_N = 4 * N_SENT + 2 * N_CTXN;   // 500000
  int* deg_s2s_out = deg + 0;
  int* deg_s2s_in  = deg + N_SENT;
  int* deg_s2c_out = deg + 2 * N_SENT;
  int* deg_s2c_in  = deg + 3 * N_SENT;
  int* deg_c2s_out = deg + 3 * N_SENT + N_CTXN;
  int* deg_c2s_in  = deg + 3 * N_SENT + 2 * N_CTXN;
  p += (size_t)DEG_N * 4;

  float* rs = (float*)p;
  float* rs_s2s_out = rs + 0;
  float* rs_s2s_in  = rs + N_SENT;
  float* rs_s2c_out = rs + 2 * N_SENT;
  float* rs_s2c_in  = rs + 3 * N_SENT;
  float* rs_c2s_out = rs + 3 * N_SENT + N_CTXN;
  float* rs_c2s_in  = rs + 3 * N_SENT + 2 * N_CTXN;
  p += (size_t)DEG_N * 4;

  float* agg  = (float*)p; p += (size_t)N_SENT * DH * 4;
  float* h_s  = (float*)p; p += (size_t)N_SENT * DH * 4;
  float* h_c  = (float*)p; p += (size_t)N_CTXN * DH * 4;
  float* h_s2 = (float*)p; p += (size_t)N_SENT * DH * 4;
  unsigned short* Wp = (unsigned short*)p;     // 6 * 16384 bf16
  const int WPN = 16384;

  float* out_f   = (float*)d_out;              // [0, 4096): document_feature
  float* h_c2out = out_f + N_GRAPHS * DH;      // [4096, ...): h_c2

  const int BT = 256;
  auto g = [](long long n, int bt) { return (unsigned)((n + bt - 1) / bt); };

  // ---- degrees ----
  zero_i32<<<g(DEG_N, BT), BT, 0, stream>>>(deg, DEG_N);
  count_deg<<<g(ne_s2s, BT), BT, 0, stream>>>(e_s2s_src, e_s2s_dst, deg_s2s_out, deg_s2s_in, ne_s2s);
  count_deg<<<g(ne_s2c, BT), BT, 0, stream>>>(e_s2c_src, e_s2c_dst, deg_s2c_out, deg_s2c_in, ne_s2c);
  count_deg<<<g(ne_c2s, BT), BT, 0, stream>>>(e_c2s_src, e_c2s_dst, deg_c2s_out, deg_c2s_in, ne_c2s);
  deg_to_rsqrt<<<g(DEG_N, BT), BT, 0, stream>>>(deg, rs, DEG_N);

  // ---- pack weights to bf16 B-fragment layout ----
  pack_w_bf16<<<g(WPN, BT), BT, 0, stream>>>(W1_s2s, Wp + 0 * WPN);
  pack_w_bf16<<<g(WPN, BT), BT, 0, stream>>>(W1_c2s, Wp + 1 * WPN);
  pack_w_bf16<<<g(WPN, BT), BT, 0, stream>>>(W1_s2c, Wp + 2 * WPN);
  pack_w_bf16<<<g(WPN, BT), BT, 0, stream>>>(W2_s2s, Wp + 3 * WPN);
  pack_w_bf16<<<g(WPN, BT), BT, 0, stream>>>(W2_c2s, Wp + 4 * WPN);
  pack_w_bf16<<<g(WPN, BT), BT, 0, stream>>>(W2_s2c, Wp + 5 * WPN);

  // ---- layer 1 ----
  // s2s -> h_s (write)
  fill_f32<<<g((long long)N_SENT * DH, BT), BT, 0, stream>>>(agg, 0.f, N_SENT * DH);
  spmm_scatter<<<g((long long)ne_s2s * 32, BT), BT, 0, stream>>>(feat_s, e_s2s_src, e_s2s_dst, rs_s2s_out, agg, ne_s2s);
  gemm_wmma<0, 0><<<N_SENT / 16, 256, 0, stream>>>(agg, rs_s2s_in, Wp + 0 * WPN, b1_s2s, h_s);
  // c2s -> h_s (accumulate + relu)
  fill_f32<<<g((long long)N_SENT * DH, BT), BT, 0, stream>>>(agg, 0.f, N_SENT * DH);
  spmm_scatter<<<g((long long)ne_c2s * 32, BT), BT, 0, stream>>>(feat_c, e_c2s_src, e_c2s_dst, rs_c2s_out, agg, ne_c2s);
  gemm_wmma<1, 1><<<N_SENT / 16, 256, 0, stream>>>(agg, rs_c2s_in, Wp + 1 * WPN, b1_c2s, h_s);
  // s2c -> h_c (write + relu)
  fill_f32<<<g((long long)N_CTXN * DH, BT), BT, 0, stream>>>(agg, 0.f, N_CTXN * DH);
  spmm_scatter<<<g((long long)ne_s2c * 32, BT), BT, 0, stream>>>(feat_s, e_s2c_src, e_s2c_dst, rs_s2c_out, agg, ne_s2c);
  gemm_wmma<0, 1><<<N_CTXN / 16, 256, 0, stream>>>(agg, rs_s2c_in, Wp + 2 * WPN, b1_s2c, h_c);

  // ---- layer 2 ----
  // s2s -> h_s2 (write)
  fill_f32<<<g((long long)N_SENT * DH, BT), BT, 0, stream>>>(agg, 0.f, N_SENT * DH);
  spmm_scatter<<<g((long long)ne_s2s * 32, BT), BT, 0, stream>>>(h_s, e_s2s_src, e_s2s_dst, rs_s2s_out, agg, ne_s2s);
  gemm_wmma<0, 0><<<N_SENT / 16, 256, 0, stream>>>(agg, rs_s2s_in, Wp + 3 * WPN, b2_s2s, h_s2);
  // c2s -> h_s2 (accumulate)
  fill_f32<<<g((long long)N_SENT * DH, BT), BT, 0, stream>>>(agg, 0.f, N_SENT * DH);
  spmm_scatter<<<g((long long)ne_c2s * 32, BT), BT, 0, stream>>>(h_c, e_c2s_src, e_c2s_dst, rs_c2s_out, agg, ne_c2s);
  gemm_wmma<1, 0><<<N_SENT / 16, 256, 0, stream>>>(agg, rs_c2s_in, Wp + 4 * WPN, b2_c2s, h_s2);
  // s2c -> h_c2 (direct to output)
  fill_f32<<<g((long long)N_CTXN * DH, BT), BT, 0, stream>>>(agg, 0.f, N_CTXN * DH);
  spmm_scatter<<<g((long long)ne_s2c * 32, BT), BT, 0, stream>>>(h_s, e_s2c_src, e_s2c_dst, rs_s2c_out, agg, ne_s2c);
  gemm_wmma<0, 0><<<N_CTXN / 16, 256, 0, stream>>>(agg, rs_s2c_in, Wp + 5 * WPN, b2_s2c, h_c2out);

  // ---- segment max over sentences -> document_feature ----
  fill_f32<<<g(N_GRAPHS * DH, BT), BT, 0, stream>>>(out_f, -__builtin_inff(), N_GRAPHS * DH);
  segmax_kernel<<<g((long long)N_SENT * DH, BT), BT, 0, stream>>>(h_s2, gid, out_f, N_SENT);
}
